// VTV_GCL_Dense_18580028522830
// MI455X (gfx1250) — compile-verified, matrix operand order checked by reading, not run on hardware
//
#include <hip/hip_runtime.h>
#include <hip/hip_bf16.h>
#include <cstdint>

// ---------------------------------------------------------------------------
// MI455X (gfx1250) implementation. wave32, WMMA f32<-f16 16x16x32.
// ---------------------------------------------------------------------------

typedef _Float16 v16h __attribute__((ext_vector_type(16)));
typedef _Float16 v8h  __attribute__((ext_vector_type(8)));
typedef float    v8f  __attribute__((ext_vector_type(8)));

#define NF 128

__device__ __forceinline__ float silu_f(float v) {
    return v / (1.f + __expf(-v));
}

// Load one 16-lane-striped WMMA operand fragment (A layout; B uses the same
// pattern against an N-major (transposed) weight array).
// p must point at element [row*ld + k0 + laneHi*8]; reads p[0..7], p[16..23].
__device__ __forceinline__ v16h ld_frag(const _Float16* p) {
    v8h lo = *(const v8h*)(p);
    v8h hi = *(const v8h*)(p + 16);
    return __builtin_shufflevector(lo, hi, 0,1,2,3,4,5,6,7,8,9,10,11,12,13,14,15);
}

__device__ __forceinline__ v8f wmma_f16(v16h a, v16h b, v8f c) {
    return __builtin_amdgcn_wmma_f32_16x16x32_f16(false, a, false, b,
                                                  (short)0, c, false, false);
}

// ---------------------------------------------------------------------------
// prep kernels
// ---------------------------------------------------------------------------
__global__ void k_scan_off(const int* __restrict__ degs, int* __restrict__ off, int B) {
    int b = blockIdx.x * blockDim.x + threadIdx.x;
    if (b >= B) return;
    int s = 0;
    for (int k = 0; k < b; ++k) s += degs[k];
    off[b] = s;
}

// W (K x N, f32, row-major) -> WT (N x K, f16, row-major)
__global__ void k_transpose_w(const float* __restrict__ W, _Float16* __restrict__ WT,
                              int K, int N) {
    int idx = blockIdx.x * blockDim.x + threadIdx.x;
    if (idx >= K * N) return;
    int k = idx / N, n = idx - k * N;
    WT[(size_t)n * K + k] = (_Float16)W[idx];
}

// Build edge-encoder input X0 = [h[row], h[col]] (f16, E x 256) and coord_diff.
__global__ __launch_bounds__(256) void k_edge_in(
    const float* __restrict__ h, const float* __restrict__ x,
    const int* __restrict__ edges, _Float16* __restrict__ X0,
    float* __restrict__ cd, int E) {
    int gid  = blockIdx.x * blockDim.x + threadIdx.x;
    int e    = gid >> 5;
    int lane = gid & 31;
    if (e >= E) return;
    int r = edges[e], c = edges[E + e];
    for (int f = lane; f < NF; f += 32) {
        X0[(size_t)e * 256 + f]      = (_Float16)h[(size_t)r * NF + f];
        X0[(size_t)e * 256 + NF + f] = (_Float16)h[(size_t)c * NF + f];
    }
    if (lane < 3) cd[(size_t)e * 3 + lane] = x[(size_t)r * 3 + lane] - x[(size_t)c * 3 + lane];
}

// ---------------------------------------------------------------------------
// generic fused 2-layer MLP: Y = silu(X@W1+b1)@W2+b2, N==128, f16 in/out,
// f32 accumulate via v_wmma_f32_16x16x32_f16. One M-tile (16 rows) per wave;
// all 8 N-tiles accumulated in registers; hidden repacked via per-wave LDS.
// Optional concat A = [XA(Ka) | XB(Kb)]; optional dense-block output mask.
// ---------------------------------------------------------------------------
__global__ __launch_bounds__(256) void k_mlp2(
    const _Float16* __restrict__ XA, int Ka,
    const _Float16* __restrict__ XB, int Kb,
    const _Float16* __restrict__ W1T, const float* __restrict__ B1,
    const _Float16* __restrict__ W2T, const float* __restrict__ B2,
    _Float16* __restrict__ Y, int Mtiles,
    const int* __restrict__ degs, int maskdense, int blockBase) {
    __shared__ __align__(16) _Float16 sh[8][16][NF];

    const int lane = threadIdx.x & 31;
    const int wv   = threadIdx.x >> 5;
    int mtile = blockIdx.x * 8 + wv;
    const bool active = (mtile < Mtiles);
    if (!active) mtile = Mtiles - 1;                 // keep wave uniform, gate stores

    const int K     = Ka + Kb;
    const int mloc  = lane & 15;
    const int hiq   = lane >> 4;
    const int klohi = hiq * 8;
    const int arow  = mtile * 16 + mloc;

    // ---- layer 1 ----
    v8f acc[8] = {};
    for (int k0 = 0; k0 < K; k0 += 32) {
        const _Float16* ap = (k0 < Ka)
            ? (XA + (size_t)arow * Ka + k0 + klohi)
            : (XB + (size_t)arow * Kb + (k0 - Ka) + klohi);
        v16h a = ld_frag(ap);
        #pragma unroll
        for (int nt = 0; nt < 8; ++nt) {
            v16h b = ld_frag(W1T + (size_t)(nt * 16 + mloc) * K + k0 + klohi);
            acc[nt] = wmma_f16(a, b, acc[nt]);
        }
    }
    // bias + silu -> f16 hidden in LDS (C layout -> row-major repack)
    #pragma unroll
    for (int nt = 0; nt < 8; ++nt) {
        float bb = B1[nt * 16 + mloc];
        #pragma unroll
        for (int r = 0; r < 8; ++r) {
            float v = acc[nt][r] + bb;
            sh[wv][r + 8 * hiq][nt * 16 + mloc] = (_Float16)silu_f(v);
        }
    }
    // ---- layer 2 (K = 128) ----
    v8f acc2[8] = {};
    for (int k0 = 0; k0 < NF; k0 += 32) {
        v16h a = ld_frag(&sh[wv][mloc][k0 + klohi]);
        #pragma unroll
        for (int nt = 0; nt < 8; ++nt) {
            v16h b = ld_frag(W2T + (size_t)(nt * 16 + mloc) * NF + k0 + klohi);
            acc2[nt] = wmma_f16(a, b, acc2[nt]);
        }
    }
    if (!active) return;
    #pragma unroll
    for (int nt = 0; nt < 8; ++nt) {
        float bb = B2[nt * 16 + mloc];
        #pragma unroll
        for (int r = 0; r < 8; ++r) {
            int m    = r + 8 * hiq;
            int prow = mtile * 16 + m;
            float v  = acc2[nt][r] + bb;
            if (maskdense) {
                int bg = blockBase + (prow >> 8);
                int ii = (prow >> 4) & 15, jj = prow & 15;
                int d  = degs[bg];
                if (ii >= d || jj >= d) v = 0.f;
            }
            Y[(size_t)prow * NF + nt * 16 + mloc] = (_Float16)v;
        }
    }
}

// ---------------------------------------------------------------------------
// nb_x (dense, masked): pos_enc(vtv) ++ signed_sqrt(ef_i * ef_j). wave/entry.
// ---------------------------------------------------------------------------
__global__ __launch_bounds__(256) void k_nbx(
    const float* __restrict__ cd, const _Float16* __restrict__ ef,
    const int* __restrict__ degs, const int* __restrict__ off,
    _Float16* __restrict__ nbx, int blockBase, int nBlk) {
    int gid  = blockIdx.x * blockDim.x + threadIdx.x;
    int idx  = gid >> 5;              // chunk-local dense entry
    int lane = gid & 31;
    if (idx >= nBlk * 256) return;
    int b = blockBase + (idx >> 8);
    int ij = idx & 255, i = ij >> 4, j = ij & 15;
    int d = degs[b];
    _Float16* dst = nbx + (size_t)idx * 192;
    if (i >= d || j >= d) {
        for (int c = lane; c < 192; c += 32) dst[c] = (_Float16)0.f;
        return;
    }
    int ei = off[b] + i, ej = off[b] + j;
    float vtv = cd[(size_t)ei * 3 + 0] * cd[(size_t)ej * 3 + 0]
              + cd[(size_t)ei * 3 + 1] * cd[(size_t)ej * 3 + 1]
              + cd[(size_t)ei * 3 + 2] * cd[(size_t)ej * 3 + 2];
    const float ln_n = 9.210340371976184f;           // log(10000)
    for (int c = lane; c < 64; c += 32) {
        int k = c >> 1;
        float div = __expf(ln_n * (float)k * (1.f / 32.f));
        float ang = vtv * 32.f / div;
        dst[c] = (_Float16)((c & 1) ? cosf(ang) : sinf(ang));
    }
    for (int c = lane; c < NF; c += 32) {
        float t = (float)ef[(size_t)ei * NF + c] * (float)ef[(size_t)ej * NF + c];
        dst[64 + c] = (_Float16)copysignf(sqrtf(fabsf(t)), t);
    }
}

// ---------------------------------------------------------------------------
// mult[b,i,j,f] = sum_k m1[b,i,k,f] * m2[b,k,j,f]  (per-feature 16^3 batched)
// one workgroup per block, 32-feature LDS double slabs. ~0.5 GFLOP total: VALU.
// ---------------------------------------------------------------------------
__global__ __launch_bounds__(256) void k_mult(
    const _Float16* __restrict__ m1, const _Float16* __restrict__ m2,
    _Float16* __restrict__ mult, int nBlk) {
    __shared__ __align__(16) _Float16 s1[256][32];
    __shared__ __align__(16) _Float16 s2[256][32];
    int b = blockIdx.x;
    if (b >= nBlk) return;
    int tid = threadIdx.x;
    int i = tid >> 4, j = tid & 15;
    for (int fc = 0; fc < 4; ++fc) {
        __syncthreads();
        size_t base = ((size_t)b * 256 + tid) * NF + fc * 32;
        #pragma unroll
        for (int q = 0; q < 4; ++q) {
            *(v8h*)&s1[tid][q * 8] = *(const v8h*)(m1 + base + q * 8);
            *(v8h*)&s2[tid][q * 8] = *(const v8h*)(m2 + base + q * 8);
        }
        __syncthreads();
        float acc[32];
        #pragma unroll
        for (int f = 0; f < 32; ++f) acc[f] = 0.f;
        for (int k = 0; k < 16; ++k) {
            #pragma unroll
            for (int f = 0; f < 32; ++f)
                acc[f] += (float)s1[i * 16 + k][f] * (float)s2[k * 16 + j][f];
        }
        #pragma unroll
        for (int f = 0; f < 32; ++f)
            mult[((size_t)b * 256 + tid) * NF + fc * 32 + f] = (_Float16)acc[f];
    }
}

// ---------------------------------------------------------------------------
// IGN 2->1 readout + node_dec + coord head. One workgroup per block.
//   feat(16x640) in LDS -> t1 = feat @ ign_w + b (WMMA, K=640) * rowvalid
//   t0 = sum rows; h_new = h + node_dec(t0); coord weights per edge; x_new.
// ---------------------------------------------------------------------------
__global__ __launch_bounds__(256) void k_ign(
    const _Float16* __restrict__ d1,               // chunk-local (nBlk*256*128)
    const _Float16* __restrict__ ignWT, const float* __restrict__ ignB,
    const _Float16* __restrict__ ndW1T, const float* __restrict__ ndB1,
    const _Float16* __restrict__ ndW2T, const float* __restrict__ ndB2,
    const _Float16* __restrict__ cW1T,  const float* __restrict__ cB1,
    const float* __restrict__ cW2,
    const float* __restrict__ h, const float* __restrict__ x,
    const float* __restrict__ cd, const int* __restrict__ degs,
    const int* __restrict__ off,
    float* __restrict__ hOut, float* __restrict__ xOut, int blockBase) {
    __shared__ __align__(16) _Float16 feat[16][640];
    __shared__ float t1s[16][NF];
    __shared__ float t0s[NF];
    __shared__ float hmid[NF];
    __shared__ float wpart[16][16];
    __shared__ float transs[16][3];

    int bl = blockIdx.x, b = blockBase + bl;
    int tid = threadIdx.x;
    int d = degs[b];

    // ---- phase 1: diag / rowsum / colsum / diagsum / totalsum -> feat ----
    if (tid < NF) {
        int f = tid;
        float cs[16];
        #pragma unroll
        for (int j = 0; j < 16; ++j) cs[j] = 0.f;
        float dsum = 0.f, total = 0.f;
        const _Float16* base = d1 + ((size_t)bl * 256) * NF + f;
        for (int i = 0; i < 16; ++i) {
            float rs = 0.f, dg = 0.f;
            for (int j = 0; j < 16; ++j) {
                float v = (float)base[(size_t)(i * 16 + j) * NF];
                rs += v; cs[j] += v;
                if (j == i) dg = v;
            }
            feat[i][f]       = (_Float16)dg;
            feat[i][128 + f] = (_Float16)rs;
            dsum += dg; total += rs;
        }
        for (int j = 0; j < 16; ++j) feat[j][256 + f] = (_Float16)cs[j];
        for (int i = 0; i < 16; ++i) {
            feat[i][384 + f] = (_Float16)dsum;
            feat[i][512 + f] = (_Float16)total;
        }
    }
    __syncthreads();

    // ---- phase 2: t1 = feat @ ign_w + b (WMMA, M=16, K=640, wave = N-tile) ----
    {
        int lane = tid & 31, wv = tid >> 5;
        int mloc = lane & 15, hiq = lane >> 4, klohi = hiq * 8;
        v8f acc = {};
        for (int k0 = 0; k0 < 640; k0 += 32) {
            v16h a  = ld_frag(&feat[mloc][k0 + klohi]);
            v16h bf = ld_frag(ignWT + (size_t)(wv * 16 + mloc) * 640 + k0 + klohi);
            acc = wmma_f16(a, bf, acc);
        }
        float bb = ignB[wv * 16 + mloc];
        #pragma unroll
        for (int r = 0; r < 8; ++r) {
            int m = r + 8 * hiq;
            float v = acc[r] + bb;
            t1s[m][wv * 16 + mloc] = (m < d) ? v : 0.f;
        }
    }
    __syncthreads();

    // ---- phase 3: t0 ----
    if (tid < NF) {
        float s = 0.f;
        for (int i = 0; i < 16; ++i) s += t1s[i][tid];
        t0s[tid] = s;
    }
    __syncthreads();

    // ---- phase 4a: coord hidden partials (all threads): i=tid>>4, 8 outs each
    {
        int i = tid >> 4, og = tid & 15;
        float part = 0.f;
        #pragma unroll
        for (int oc = 0; oc < 8; ++oc) {
            int o = og * 8 + oc;
            float a = cB1[o];
            const _Float16* w = cW1T + (size_t)o * NF;
            for (int f = 0; f < NF; ++f) a += t1s[i][f] * (float)w[f];
            part += silu_f(a) * cW2[o];
        }
        wpart[i][og] = part;
    }
    // ---- phase 4b: node_dec layer 1 ----
    if (tid < NF) {
        int o = tid;
        float a = ndB1[o];
        const _Float16* w = ndW1T + (size_t)o * NF;
        for (int f = 0; f < NF; ++f) a += t0s[f] * (float)w[f];
        hmid[o] = silu_f(a);
    }
    __syncthreads();

    // ---- phase 5a: per-edge translation ----
    if (tid < 16) {
        int i = tid;
        float w = 0.f;
        for (int og = 0; og < 16; ++og) w += wpart[i][og];
        if (i < d) {
            int e = off[b] + i;
            #pragma unroll
            for (int c = 0; c < 3; ++c) transs[i][c] = cd[(size_t)e * 3 + c] * w;
        } else {
            #pragma unroll
            for (int c = 0; c < 3; ++c) transs[i][c] = 0.f;
        }
    }
    // ---- phase 5b: node_dec layer 2 + residual ----
    if (tid < NF) {
        int o = tid;
        float a = ndB2[o];
        const _Float16* w = ndW2T + (size_t)o * NF;
        for (int f = 0; f < NF; ++f) a += hmid[f] * (float)w[f];
        hOut[(size_t)b * NF + o] = h[(size_t)b * NF + o] + a;
    }
    __syncthreads();

    // ---- phase 6: x_new = x + mean(trans) ----
    if (tid < 3) {
        float s = 0.f;
        for (int i = 0; i < 16; ++i) s += transs[i][tid];
        xOut[(size_t)b * 3 + tid] = x[(size_t)b * 3 + tid] + s / (float)d;
    }
}

// ---------------------------------------------------------------------------
// launcher
// ---------------------------------------------------------------------------
extern "C" void kernel_launch(void* const* d_in, const int* in_sizes, int n_in,
                              void* d_out, int out_size, void* d_ws, size_t ws_size,
                              hipStream_t stream) {
    (void)n_in; (void)out_size; (void)ws_size;
    // Input layout: insertion order of setup_inputs() with params flattened.
    const float* h     = (const float*)d_in[0];
    const float* x     = (const float*)d_in[1];
    const float* eattr = (const float*)d_in[2];
    const float* eeW1 = (const float*)d_in[3];  const float* eeB1 = (const float*)d_in[4];
    const float* eeW2 = (const float*)d_in[5];  const float* eeB2 = (const float*)d_in[6];
    const float* p1W1 = (const float*)d_in[7];  const float* p1B1 = (const float*)d_in[8];
    const float* p1W2 = (const float*)d_in[9];  const float* p1B2 = (const float*)d_in[10];
    const float* p2W1 = (const float*)d_in[11]; const float* p2B1 = (const float*)d_in[12];
    const float* p2W2 = (const float*)d_in[13]; const float* p2B2 = (const float*)d_in[14];
    const float* p3W1 = (const float*)d_in[15]; const float* p3B1 = (const float*)d_in[16];
    const float* p3W2 = (const float*)d_in[17]; const float* p3B2 = (const float*)d_in[18];
    const float* ignW = (const float*)d_in[19]; const float* ignB = (const float*)d_in[20];
    const float* ndW1 = (const float*)d_in[21]; const float* ndB1 = (const float*)d_in[22];
    const float* ndW2 = (const float*)d_in[23]; const float* ndB2 = (const float*)d_in[24];
    const float* cW1  = (const float*)d_in[25]; const float* cB1  = (const float*)d_in[26];
    const float* cW2  = (const float*)d_in[27];
    const int* edges  = (const int*)d_in[28];
    const int* degs   = (const int*)d_in[30];

    const int E = in_sizes[28] / 2;
    const int B = in_sizes[30];

    // workspace carve (256B aligned)
    char* wp = (char*)d_ws;
    auto alloc = [&](size_t bytes) -> void* {
        void* r = (void*)wp;
        wp += (bytes + 255) & ~(size_t)255;
        return r;
    };
    int*      off   = (int*)alloc((size_t)B * 4);
    _Float16* eW1T  = (_Float16*)alloc((size_t)128 * 256 * 2);
    _Float16* eW2T  = (_Float16*)alloc((size_t)128 * 128 * 2);
    _Float16* m1W1T = (_Float16*)alloc((size_t)128 * 192 * 2);
    _Float16* m1W2T = (_Float16*)alloc((size_t)128 * 128 * 2);
    _Float16* m2W1T = (_Float16*)alloc((size_t)128 * 192 * 2);
    _Float16* m2W2T = (_Float16*)alloc((size_t)128 * 128 * 2);
    _Float16* m3W1T = (_Float16*)alloc((size_t)128 * 320 * 2);
    _Float16* m3W2T = (_Float16*)alloc((size_t)128 * 128 * 2);
    _Float16* ignWT = (_Float16*)alloc((size_t)128 * 640 * 2);
    _Float16* ndW1T = (_Float16*)alloc((size_t)128 * 128 * 2);
    _Float16* ndW2T = (_Float16*)alloc((size_t)128 * 128 * 2);
    _Float16* cW1T  = (_Float16*)alloc((size_t)128 * 128 * 2);
    _Float16* X0    = (_Float16*)alloc((size_t)E * 256 * 2);
    float*    cd    = (float*)alloc((size_t)E * 3 * 4);
    _Float16* ef    = (_Float16*)alloc((size_t)E * NF * 2);

    const int CHUNK = 128;
    _Float16* nbx  = (_Float16*)alloc((size_t)CHUNK * 256 * 192 * 2);
    _Float16* m1b  = (_Float16*)alloc((size_t)CHUNK * 256 * NF * 2);
    _Float16* m2b  = (_Float16*)alloc((size_t)CHUNK * 256 * NF * 2);
    _Float16* mlb  = (_Float16*)alloc((size_t)CHUNK * 256 * NF * 2);
    _Float16* d1b  = (_Float16*)alloc((size_t)CHUNK * 256 * NF * 2);

    float* hOut = (float*)d_out;
    float* xOut = hOut + (size_t)B * NF;
    float* aOut = xOut + (size_t)B * 3;

    // prep: prefix offsets + weight transposes (f32 -> f16 N-major)
    k_scan_off<<<(B + 1023) / 1024, 1024, 0, stream>>>(degs, off, B);
    auto T = [&](const float* W, _Float16* WT, int K, int Nn) {
        int n = K * Nn;
        k_transpose_w<<<(n + 255) / 256, 256, 0, stream>>>(W, WT, K, Nn);
    };
    T(eeW1, eW1T, 256, 128);  T(eeW2, eW2T, 128, 128);
    T(p1W1, m1W1T, 192, 128); T(p1W2, m1W2T, 128, 128);
    T(p2W1, m2W1T, 192, 128); T(p2W2, m2W2T, 128, 128);
    T(p3W1, m3W1T, 320, 128); T(p3W2, m3W2T, 128, 128);
    T(ignW, ignWT, 640, 128);
    T(ndW1, ndW1T, 128, 128); T(ndW2, ndW2T, 128, 128);
    T(cW1,  cW1T,  128, 128);

    // edge encoder
    k_edge_in<<<(E * 32 + 255) / 256, 256, 0, stream>>>(h, x, edges, X0, cd, E);
    {
        int Mt = (E + 15) / 16;
        k_mlp2<<<(Mt + 7) / 8, 256, 0, stream>>>(X0, 256, nullptr, 0,
                                                 eW1T, eeB1, eW2T, eeB2,
                                                 ef, Mt, nullptr, 0, 0);
    }

    // per-block PPGN pipeline, chunked (workspace + L2 residency)
    for (int bb = 0; bb < B; bb += CHUNK) {
        int cb = (B - bb < CHUNK) ? (B - bb) : CHUNK;
        int Mt = cb * 16;  // cb*256 rows / 16
        k_nbx<<<cb * 32, 256, 0, stream>>>(cd, ef, degs, off, nbx, bb, cb);
        k_mlp2<<<(Mt + 7) / 8, 256, 0, stream>>>(nbx, 192, nullptr, 0,
                                                 m1W1T, p1B1, m1W2T, p1B2,
                                                 m1b, Mt, degs, 1, bb);
        k_mlp2<<<(Mt + 7) / 8, 256, 0, stream>>>(nbx, 192, nullptr, 0,
                                                 m2W1T, p2B1, m2W2T, p2B2,
                                                 m2b, Mt, degs, 1, bb);
        k_mult<<<cb, 256, 0, stream>>>(m1b, m2b, mlb, cb);
        k_mlp2<<<(Mt + 7) / 8, 256, 0, stream>>>(nbx, 192, mlb, 128,
                                                 m3W1T, p3B1, m3W2T, p3B2,
                                                 d1b, Mt, degs, 1, bb);
        k_ign<<<cb, 256, 0, stream>>>(d1b, ignWT, ignB, ndW1T, ndB1, ndW2T, ndB2,
                                      cW1T, cB1, cW2, h, x, cd, degs, off,
                                      hOut, xOut, bb);
    }

    // edge_attr passthrough (3rd tuple element of the reference output)
    hipMemcpyAsync(aOut, eattr, (size_t)in_sizes[2] * sizeof(float),
                   hipMemcpyDeviceToDevice, stream);
}
